// MODEL_26414048870431
// MI455X (gfx1250) — compile-verified, matrix operand order checked by reading, not run on hardware
//
#include <hip/hip_runtime.h>

// ---------------------------------------------------------------------------
// GIKT-style model on MI455X (gfx1250, wave32, WMMA).
//   1. f32->bf16 converts (+ E transpose)
//   2. kc_Wh (f32, also stored K-transposed bf16) ; q_kc ; v1 ; p_ex
//   3. ex_Eh = ex_embed @ E[h]                     (WMMA bf16)
//   4. attention: online softmax + att @ kc_Wh     (WMMA bf16)
//   5. elu([new_kc | new_kc*ex_Eh] @ rd_w.T + rd_b)(WMMA bf16)
//   6. head-interleave combine -> exercise embedding
//   7. persistent LSTM: 16 WGs x 16 batch, 500 steps, weight B-fragments
//      in registers, h/c in LDS, x gathered per step (WMMA bf16)
//   8. fused MLP head: gather + fc1(WMMA) + relu + fc2 + sigmoid + BCE
// All fragment loads are 32-bit (element pairs are contiguous in K).
// ---------------------------------------------------------------------------

#define N_EX   10000
#define N_KC   500
#define D_DIM  128
#define HID    128
#define GATES  512
#define BATCH  256
#define SEQ    500
#define R_OUT  (BATCH * (SEQ - 1))   // 127744
#define NEGV   (-9.0e15f)

typedef __bf16 bf16;
typedef __attribute__((ext_vector_type(16))) __bf16        v16bf;
typedef __attribute__((ext_vector_type(8)))  float         v8f;
typedef __attribute__((ext_vector_type(8)))  unsigned int  v8ui;

union frag_cv { v8ui u; v16bf b; };
union bq4     { bf16 b[4]; uint2 u; };

// ISA 7.12.2: 16-bit A 16x32 fragment: lane L holds row M=L&15, hi=L>>4.
// Element t -> K = kmap16(t,hi); pairs (2p,2p+1) are contiguous K.
__device__ __forceinline__ int kmap16(int t, int hi) {
  int v = t >> 1;
  int base = (v < 4) ? (v << 1) : (16 + ((v - 4) << 1));
  return base + (t & 1) + (hi << 3);
}
// starting (even) K of element pair p
__device__ __forceinline__ int pmap16(int p, int hi) {
  int base = (p < 4) ? (p << 1) : (16 + ((p - 4) << 1));
  return base + (hi << 3);
}

// Fragment whose K-run is contiguous at `ptr` (4B-aligned, k0 even):
// eight 32-bit loads, no packing VALU.
__device__ __forceinline__ v16bf frag_row(const bf16* ptr, int k0, int hi) {
  frag_cv cv;
  const unsigned int* p32 = (const unsigned int*)ptr;
#pragma unroll
  for (int p = 0; p < 8; ++p) cv.u[p] = p32[(k0 + pmap16(p, hi)) >> 1];
  return cv.b;
}

__device__ __forceinline__ v8f wmma_bf16(v16bf a, v16bf b, v8f c) {
  return __builtin_amdgcn_wmma_f32_16x16x32_bf16(false, a, false, b,
                                                 (short)0, c, false, false);
}

__device__ __forceinline__ float sigmf(float x) { return 1.f / (1.f + __expf(-x)); }

// ---------------------------------------------------------------- converts
__global__ void k_cvt_bf16(const float* __restrict__ s, bf16* __restrict__ d, int n) {
  int i = blockIdx.x * blockDim.x + threadIdx.x;
  if (i < n) d[i] = (bf16)s[i];
}

// Et[h][n][k] = E[h][k][n]  (so WMMA B-fragments are K-contiguous)
__global__ void k_trE(const float* __restrict__ E, bf16* __restrict__ Et) {
  int i = blockIdx.x * blockDim.x + threadIdx.x;
  if (i >= 2 * D_DIM * D_DIM) return;
  int k = i & 127, n = (i >> 7) & 127, h = i >> 14;
  Et[i] = (bf16)E[h * D_DIM * D_DIM + k * D_DIM + n];
}

// --------------------------------------------------- kc_Wh (small, f32 exact)
// kcWh[h][j][d] (f32, for q_kc) and kcWhT[h][d][j] (bf16, K-contiguous B)
__global__ void k_kcwh(const float* __restrict__ kc, const float* __restrict__ W1,
                       float* __restrict__ kcWh, bf16* __restrict__ kcWhT) {
  int i = blockIdx.x * blockDim.x + threadIdx.x;
  if (i >= 2 * N_KC * D_DIM) return;
  int d = i & 127, j = (i >> 7) % N_KC, h = i / (N_KC * D_DIM);
  const float* wcol = W1 + h * D_DIM * D_DIM + d;
  const float* krow = kc + j * D_DIM;
  float s = 0.f;
  for (int k = 0; k < D_DIM; ++k) s += krow[k] * wcol[k * D_DIM];
  kcWh[i] = s;
  kcWhT[h * D_DIM * N_KC + d * N_KC + j] = (bf16)s;
}

// v1[h] = W1[h] @ a_att[h][:D]   (p_ex = ex_embed @ v1 avoids full ex_Wh)
__global__ void k_v1(const float* __restrict__ W1, const float* __restrict__ a_att,
                     float* __restrict__ v1) {
  int i = blockIdx.x * blockDim.x + threadIdx.x;
  if (i >= 2 * D_DIM) return;
  int h = i >> 7, r = i & 127;
  const float* wrow = W1 + h * D_DIM * D_DIM + r * D_DIM;
  const float* av = a_att + h * 2 * D_DIM;
  float s = 0.f;
  for (int d = 0; d < D_DIM; ++d) s += wrow[d] * av[d];
  v1[i] = s;
}

__global__ void k_qkc(const float* __restrict__ kcWh, const float* __restrict__ a_att,
                      float* __restrict__ q) {
  int i = blockIdx.x * blockDim.x + threadIdx.x;
  if (i >= 2 * N_KC) return;
  int h = i / N_KC, j = i % N_KC;
  const float* row = kcWh + (h * N_KC + j) * D_DIM;
  const float* av = a_att + h * 2 * D_DIM + D_DIM;
  float s = 0.f;
  for (int d = 0; d < D_DIM; ++d) s += row[d] * av[d];
  q[i] = s;
}

__global__ void k_pex(const float* __restrict__ ex, const float* __restrict__ v1,
                      float* __restrict__ p) {
  int i = blockIdx.x * blockDim.x + threadIdx.x;
  if (i >= 2 * N_EX) return;
  int h = i / N_EX, n = i % N_EX;
  const float* er = ex + (size_t)n * D_DIM;
  const float* vv = v1 + h * D_DIM;
  float s = 0.f;
  for (int d = 0; d < D_DIM; ++d) s += er[d] * vv[d];
  p[i] = s;
}

// ------------------------------------------ ex_Eh = ex_embed @ E[h]  (WMMA)
__global__ void k_exeh(const bf16* __restrict__ ex_bf, const bf16* __restrict__ Et,
                       float* __restrict__ exEh) {
  int rt = blockIdx.x, ct = blockIdx.y, h = blockIdx.z;
  int lane = threadIdx.x, nl = lane & 15, hi = lane >> 4;
  const bf16* arow = ex_bf + (size_t)(rt * 16 + nl) * D_DIM;
  const bf16* brow = Et + h * D_DIM * D_DIM + (ct * 16 + nl) * D_DIM;
  v8f acc = {};
#pragma unroll
  for (int kk = 0; kk < 4; ++kk) {
    v16bf a = frag_row(arow, kk * 32, hi);
    v16bf b = frag_row(brow, kk * 32, hi);
    acc = wmma_bf16(a, b, acc);
  }
  float* out = exEh + (size_t)h * N_EX * D_DIM;
#pragma unroll
  for (int r = 0; r < 8; ++r)
    out[(size_t)(rt * 16 + r + 8 * hi) * D_DIM + ct * 16 + nl] = acc[r];
}

// ---------------- attention: online softmax + new_kc = att @ kc_Wh  (WMMA)
__global__ void k_att(const int* __restrict__ adj, const float* __restrict__ p_ex,
                      const float* __restrict__ q_kc, const bf16* __restrict__ kcWhT,
                      float* __restrict__ newkc) {
  int rt = blockIdx.x, h = blockIdx.y;
  int lane = threadIdx.x, nl = lane & 15, hi = lane >> 4;
  int row = rt * 16 + nl;
  float p = p_ex[h * N_EX + row];
  const int* arow = adj + (size_t)row * N_KC;
  const float* q = q_kc + h * N_KC;
  __builtin_prefetch(arow, 0, 0);

  // pass 1: online max/sum over masked leaky-relu logits (NEG where adj==0,
  // matching jnp.where incl. the all-masked -> uniform softmax case)
  float m = -INFINITY, s = 0.f;
  for (int j = hi; j < N_KC; j += 2) {
    float z = p + q[j];
    float e = (arow[j] > 0) ? (z > 0.f ? z : 0.2f * z) : NEGV;
    if (e > m) { s = s * __expf(m - e) + 1.f; m = e; }
    else       { s += __expf(e - m); }
  }
  float om = __shfl_xor(m, 16, 32);
  float os = __shfl_xor(s, 16, 32);
  float M = fmaxf(m, om);
  float S = s * __expf(m - M) + os * __expf(om - M);
  float invS = 1.f / S;

  // pass 2: rebuild attention weights per fragment, WMMA against kc_Wh^T
  const bf16* kbT = kcWhT + h * D_DIM * N_KC;
  v8f acc[8];
#pragma unroll
  for (int c = 0; c < 8; ++c) acc[c] = (v8f){};
  for (int ch = 0; ch < 16; ++ch) {
    int j0 = ch * 32;
    v16bf a;
#pragma unroll
    for (int t = 0; t < 16; ++t) {
      int j = j0 + kmap16(t, hi);
      float w = 0.f;
      if (j < N_KC) {
        float z = p + q[j];
        float e = (arow[j] > 0) ? (z > 0.f ? z : 0.2f * z) : NEGV;
        w = __expf(e - M) * invS;
      }
      a[t] = (bf16)w;
    }
#pragma unroll
    for (int c = 0; c < 8; ++c) {
      frag_cv cv;
      const unsigned int* p32 = (const unsigned int*)(kbT + (c * 16 + nl) * N_KC);
#pragma unroll
      for (int pp = 0; pp < 8; ++pp) {
        int j = j0 + pmap16(pp, hi);           // even; pair fully in/out
        cv.u[pp] = (j < N_KC) ? p32[j >> 1] : 0u;
      }
      acc[c] = wmma_bf16(a, cv.b, acc[c]);
    }
  }
  float* out = newkc + (size_t)h * N_EX * D_DIM;
#pragma unroll
  for (int c = 0; c < 8; ++c)
#pragma unroll
    for (int r = 0; r < 8; ++r)
      out[(size_t)(rt * 16 + r + 8 * hi) * D_DIM + c * 16 + nl] = acc[c][r];
}

// ------- elu([new_kc | new_kc*ex_Eh] @ rd_w.T + rd_b) per head  (WMMA)
__global__ void k_rd(const float* __restrict__ newkc, const float* __restrict__ exEh,
                     const bf16* __restrict__ rdw_bf, const float* __restrict__ rd_b,
                     float* __restrict__ hout) {
  int rt = blockIdx.x, h = blockIdx.y;
  int lane = threadIdx.x, nl = lane & 15, hi = lane >> 4;
  int row = rt * 16 + nl;
  const float* nk = newkc + (size_t)h * N_EX * D_DIM + (size_t)row * D_DIM;
  const float* ee = exEh + (size_t)h * N_EX * D_DIM + (size_t)row * D_DIM;
  const bf16* Bw = rdw_bf + h * D_DIM * 2 * D_DIM;
  v8f acc[8];
#pragma unroll
  for (int c = 0; c < 8; ++c) acc[c] = (v8f){};
  for (int kk = 0; kk < 8; ++kk) {
    v16bf a;
#pragma unroll
    for (int t = 0; t < 16; ++t) {
      int k = kk * 32 + kmap16(t, hi);
      float v = (k < D_DIM) ? nk[k] : nk[k - D_DIM] * ee[k - D_DIM];
      a[t] = (bf16)v;
    }
#pragma unroll
    for (int c = 0; c < 8; ++c)
      acc[c] = wmma_bf16(a, frag_row(Bw + (c * 16 + nl) * 2 * D_DIM, kk * 32, hi),
                         acc[c]);
  }
  float* out = hout + (size_t)h * N_EX * D_DIM;
#pragma unroll
  for (int c = 0; c < 8; ++c)
#pragma unroll
    for (int r = 0; r < 8; ++r) {
      int n = c * 16 + nl;
      float x = acc[c][r] + rd_b[h * D_DIM + n];
      float y = (x > 0.f) ? x : (__expf(x) - 1.f);   // elu
      out[(size_t)(rt * 16 + r + 8 * hi) * D_DIM + n] = y;
    }
}

// reshape(N_EX,2D)->reshape(N_EX,D,2).mean(-1): pairs interleave heads
__global__ void k_combine(const float* __restrict__ hout, float* __restrict__ emb,
                          bf16* __restrict__ emb_bf) {
  int i = blockIdx.x * blockDim.x + threadIdx.x;
  if (i >= N_EX * D_DIM) return;
  int d = i & 127, n = i >> 7;
  float v;
  if (d < 64) {
    const float* h0 = hout + (size_t)n * D_DIM;
    v = 0.5f * (h0[2 * d] + h0[2 * d + 1]);
  } else {
    const float* h1 = hout + (size_t)N_EX * D_DIM + (size_t)n * D_DIM;
    v = 0.5f * (h1[2 * d - 128] + h1[2 * d - 127]);
  }
  emb[i] = v;
  emb_bf[i] = (bf16)v;
}

// ------------------------- persistent LSTM: 16 WGs x 16 batch, 500 steps
__global__ void __launch_bounds__(512) k_lstm(
    const int* __restrict__ respond, const bf16* __restrict__ emb_bf,
    const bf16* __restrict__ wih_bf, const bf16* __restrict__ whh_bf,
    const float* __restrict__ b_ih, const float* __restrict__ b_hh,
    bf16* __restrict__ h_out) {
  __shared__ bf16  xs[16 * 256];   // gathered input tile
  __shared__ bf16  hs[16 * 128];   // hidden state (WMMA A source)
  __shared__ float gs[16 * 512];   // gate pre-activations
  __shared__ float cs[16 * 128];   // cell state
  __shared__ float bsum[GATES];    // b_ih + b_hh

  int tid = threadIdx.x;
  int w = tid >> 5, lane = tid & 31, nl = lane & 15, hi = lane >> 4;
  int bb = blockIdx.x * 16;

  // Weight B-fragments live in registers across all 500 steps.
  v16bf bi[8][2], bh[4][2];
#pragma unroll
  for (int kk = 0; kk < 8; ++kk)
#pragma unroll
    for (int c = 0; c < 2; ++c) {
      int g = w * 32 + c * 16 + nl;
      bi[kk][c] = frag_row(wih_bf + g * 256, kk * 32, hi);
      if (kk < 4) bh[kk][c] = frag_row(whh_bf + g * 128, kk * 32, hi);
    }
  for (int i = tid; i < 16 * 128; i += 512) { hs[i] = (bf16)0.f; cs[i] = 0.f; }
  if (tid < GATES) bsum[tid] = b_ih[tid] + b_hh[tid];
  __syncthreads();

  for (int t = 0; t < SEQ; ++t) {
    // gather x_t (resp0: zeros / [0|emb] / [emb|0]), 8 bf16 per thread (b128)
    {
      int row = tid >> 5, col0 = (tid & 31) * 8;
      int idx = respond[(bb + row) * SEQ + t];
      uint4 val = {0u, 0u, 0u, 0u};
      if (idx > 0) {
        if (idx <= N_EX) {
          if (col0 >= 128)
            val = *(const uint4*)(emb_bf + (size_t)(idx - 1) * 128 + col0 - 128);
        } else {
          if (col0 < 128)
            val = *(const uint4*)(emb_bf + (size_t)(idx - N_EX - 1) * 128 + col0);
        }
      }
      *(uint4*)(xs + row * 256 + col0) = val;
    }
    __syncthreads();

    // gates = x @ w_ih.T + h @ w_hh.T
    v8f a0 = {}, a1 = {};
#pragma unroll
    for (int kk = 0; kk < 8; ++kk) {
      v16bf a = frag_row(xs + nl * 256, kk * 32, hi);
      a0 = wmma_bf16(a, bi[kk][0], a0);
      a1 = wmma_bf16(a, bi[kk][1], a1);
    }
#pragma unroll
    for (int kk = 0; kk < 4; ++kk) {
      v16bf a = frag_row(hs + nl * 128, kk * 32, hi);
      a0 = wmma_bf16(a, bh[kk][0], a0);
      a1 = wmma_bf16(a, bh[kk][1], a1);
    }
#pragma unroll
    for (int r = 0; r < 8; ++r) {
      int m = r + 8 * hi;
      gs[m * 512 + w * 32 + nl]      = a0[r];
      gs[m * 512 + w * 32 + 16 + nl] = a1[r];
    }
    __syncthreads();

    // elementwise cell/hidden update: 4 contiguous units per thread
    {
      int i0 = tid * 4;
      int row = i0 >> 7, u0 = i0 & 127;
      const float* g = gs + row * 512;
      float4 c4 = *(const float4*)(cs + i0);
      float cold[4] = {c4.x, c4.y, c4.z, c4.w};
      float cnew[4];
      bq4 hq;
#pragma unroll
      for (int qi = 0; qi < 4; ++qi) {
        int u = u0 + qi;
        float gi = g[u]       + bsum[u];
        float gf = g[128 + u] + bsum[128 + u];
        float gg = g[256 + u] + bsum[256 + u];
        float go = g[384 + u] + bsum[384 + u];
        float cc = sigmf(gf) * cold[qi] + sigmf(gi) * tanhf(gg);
        float hh = sigmf(go) * tanhf(cc);
        cnew[qi] = cc;
        hq.b[qi] = (bf16)hh;
      }
      *(float4*)(cs + i0) = make_float4(cnew[0], cnew[1], cnew[2], cnew[3]);
      *(uint2*)(hs + i0) = hq.u;
      if (t < SEQ - 1)
        *(uint2*)(h_out + ((size_t)(bb + row) * (SEQ - 1) + t) * 128 + u0) = hq.u;
    }
    __syncthreads();
  }
}

// ------- fused head: [learn | assess] @ fc1.T -> relu -> fc2 -> BCE (WMMA)
__global__ void k_mlp(const bf16* __restrict__ h_out, const bf16* __restrict__ emb_bf,
                      const int* __restrict__ exd, const bf16* __restrict__ fc1_bf,
                      const float* __restrict__ fc1_b, const float* __restrict__ fc2_w,
                      const float* __restrict__ fc2_b, const float* __restrict__ target,
                      float* __restrict__ out, float* __restrict__ lacc) {
  int lane = threadIdx.x, nl = lane & 15, hi = lane >> 4;
  int rr = blockIdx.x * 16 + nl;                   // row = b*499 + t
  int b = rr / (SEQ - 1), tt = rr % (SEQ - 1);
  const bf16* lrow = h_out + (size_t)rr * 128;
  int idx = exd[b * SEQ + tt + 1];
  const bf16* arow = (idx > 0) ? (emb_bf + (size_t)(idx - 1) * 128) : (const bf16*)nullptr;

  v8f acc[8];
#pragma unroll
  for (int c = 0; c < 8; ++c) acc[c] = (v8f){};
#pragma unroll
  for (int kk = 0; kk < 8; ++kk) {
    v16bf a;
    if (kk < 4) a = frag_row(lrow, kk * 32, hi);                 // learn half
    else if (arow) a = frag_row(arow, kk * 32 - 128, hi);        // assess half
    else { frag_cv z; z.u = (v8ui){}; a = z.b; }
#pragma unroll
    for (int c = 0; c < 8; ++c)
      acc[c] = wmma_bf16(a, frag_row(fc1_bf + (c * 16 + nl) * 256, kk * 32, hi),
                         acc[c]);
  }
  // relu + fc2 dot (lane holds n = c*16+nl columns of rows m = r+8*hi)
  float part[8];
#pragma unroll
  for (int r = 0; r < 8; ++r) {
    float s = 0.f;
#pragma unroll
    for (int c = 0; c < 8; ++c) {
      float y = fmaxf(acc[c][r] + fc1_b[c * 16 + nl], 0.f);
      s += y * fc2_w[c * 16 + nl];
    }
    part[r] = s;
  }
#pragma unroll
  for (int off = 1; off <= 8; off <<= 1)
#pragma unroll
    for (int r = 0; r < 8; ++r) part[r] += __shfl_xor(part[r], off, 32);

  if (nl == 0) {
    float lsum = 0.f;
    float f2b = fc2_b[0];
#pragma unroll
    for (int r = 0; r < 8; ++r) {
      int row = blockIdx.x * 16 + r + 8 * hi;
      float pred = part[r] + f2b;
      out[1 + row] = sigmf(pred);
      float tv = target[row];
      out[1 + R_OUT + row] = tv;
      lsum += fmaxf(pred, 0.f) - pred * tv + log1pf(__expf(-fabsf(pred)));
    }
    atomicAdd(lacc, lsum);
  }
}

__global__ void k_zero(float* p) { if (threadIdx.x == 0 && blockIdx.x == 0) *p = 0.f; }
__global__ void k_final(const float* lacc, float* out) {
  if (threadIdx.x == 0 && blockIdx.x == 0) out[0] = lacc[0] / (float)R_OUT;
}

// ---------------------------------------------------------------------------
extern "C" void kernel_launch(void* const* d_in, const int* in_sizes, int n_in,
                              void* d_out, int out_size, void* d_ws, size_t ws_size,
                              hipStream_t stream) {
  (void)in_sizes; (void)n_in; (void)out_size; (void)ws_size;
  const int*   adj    = (const int*)d_in[0];
  const int*   exd    = (const int*)d_in[2];
  const int*   resp   = (const int*)d_in[3];
  const float* target = (const float*)d_in[4];
  const float* ex_e   = (const float*)d_in[5];
  const float* kc_e   = (const float*)d_in[6];
  const float* W1     = (const float*)d_in[7];
  const float* a_att  = (const float*)d_in[8];
  const float* E      = (const float*)d_in[9];
  const float* rd_w   = (const float*)d_in[10];
  const float* rd_b   = (const float*)d_in[11];
  const float* w_ih   = (const float*)d_in[12];
  const float* w_hh   = (const float*)d_in[13];
  const float* b_ih   = (const float*)d_in[14];
  const float* b_hh   = (const float*)d_in[15];
  const float* fc1_w  = (const float*)d_in[16];
  const float* fc1_b  = (const float*)d_in[17];
  const float* fc2_w  = (const float*)d_in[18];
  const float* fc2_b  = (const float*)d_in[19];
  float* out = (float*)d_out;

  // ---- bump allocator over d_ws ----
  char* base = (char*)d_ws;
  size_t off = 0;
  auto alloc = [&](size_t nbytes) -> void* {
    off = (off + 255) & ~(size_t)255;
    void* p = base + off;
    off += nbytes;
    return p;
  };
  bf16*  ex_bf   = (bf16*)alloc((size_t)N_EX * D_DIM * 2);
  bf16*  Et_bf   = (bf16*)alloc(2 * D_DIM * D_DIM * 2);
  bf16*  rdw_bf  = (bf16*)alloc(2 * D_DIM * 2 * D_DIM * 2);
  bf16*  wih_bf  = (bf16*)alloc(GATES * 256 * 2);
  bf16*  whh_bf  = (bf16*)alloc(GATES * 128 * 2);
  bf16*  fc1_bf  = (bf16*)alloc(D_DIM * 256 * 2);
  float* kcWh    = (float*)alloc(2 * N_KC * D_DIM * 4);
  bf16*  kcWhT   = (bf16*)alloc(2 * D_DIM * N_KC * 2);
  float* v1      = (float*)alloc(2 * D_DIM * 4);
  float* q_kc    = (float*)alloc(2 * N_KC * 4);
  float* p_ex    = (float*)alloc(2 * N_EX * 4);
  float* exEh    = (float*)alloc((size_t)2 * N_EX * D_DIM * 4);
  float* newkc   = (float*)alloc((size_t)2 * N_EX * D_DIM * 4);
  float* hout    = (float*)alloc((size_t)2 * N_EX * D_DIM * 4);
  float* emb     = (float*)alloc((size_t)N_EX * D_DIM * 4);
  bf16*  emb_bf  = (bf16*)alloc((size_t)N_EX * D_DIM * 2);
  bf16*  h_lstm  = (bf16*)alloc((size_t)R_OUT * 128 * 2);
  float* lacc    = (float*)alloc(256);

  auto cvt = [&](const float* s, bf16* d, int n) {
    k_cvt_bf16<<<(n + 255) / 256, 256, 0, stream>>>(s, d, n);
  };
  // 1. converts (+ E transpose)
  cvt(ex_e,  ex_bf,  N_EX * D_DIM);
  cvt(rd_w,  rdw_bf, 2 * D_DIM * 2 * D_DIM);
  cvt(w_ih,  wih_bf, GATES * 256);
  cvt(w_hh,  whh_bf, GATES * 128);
  cvt(fc1_w, fc1_bf, D_DIM * 256);
  k_trE<<<(2 * D_DIM * D_DIM + 255) / 256, 256, 0, stream>>>(E, Et_bf);
  // 2. small projections
  k_kcwh<<<(2 * N_KC * D_DIM + 255) / 256, 256, 0, stream>>>(kc_e, W1, kcWh, kcWhT);
  k_v1<<<1, 256, 0, stream>>>(W1, a_att, v1);
  k_qkc<<<(2 * N_KC + 255) / 256, 256, 0, stream>>>(kcWh, a_att, q_kc);
  k_pex<<<(2 * N_EX + 255) / 256, 256, 0, stream>>>(ex_e, v1, p_ex);
  // 3. ex_Eh (WMMA)
  k_exeh<<<dim3(N_EX / 16, D_DIM / 16, 2), 32, 0, stream>>>(ex_bf, Et_bf, exEh);
  // 4. attention + aggregate (WMMA)
  k_att<<<dim3(N_EX / 16, 2), 32, 0, stream>>>(adj, p_ex, q_kc, kcWhT, newkc);
  // 5. rd projection + elu (WMMA)
  k_rd<<<dim3(N_EX / 16, 2), 32, 0, stream>>>(newkc, exEh, rdw_bf, rd_b, hout);
  // 6. head combine
  k_combine<<<(N_EX * D_DIM + 255) / 256, 256, 0, stream>>>(hout, emb, emb_bf);
  // 7. persistent LSTM (WMMA, weights in registers)
  k_lstm<<<BATCH / 16, 512, 0, stream>>>(resp, emb_bf, wih_bf, whh_bf, b_ih, b_hh, h_lstm);
  // 8. fused MLP head + loss
  k_zero<<<1, 1, 0, stream>>>(lacc);
  k_mlp<<<R_OUT / 16, 32, 0, stream>>>(h_lstm, emb_bf, exd, fc1_bf, fc1_b,
                                       fc2_w, fc2_b, target, out, lacc);
  k_final<<<1, 1, 0, stream>>>(lacc, out);
}
